// bAbI_dnc_4346506904350
// MI455X (gfx1250) — matrix operand order repaired and checked by reading, not compiled, not run on hardware
//
#include <hip/hip_runtime.h>
#include <hip/hip_bf16.h>
#include <math.h>

// ---------------- problem constants ----------------
static constexpr int Bc  = 32;     // batch
static constexpr int T_  = 256;    // time
static constexpr int V_  = 10000;  // vocab
static constexpr int E_  = 256;    // embed
static constexpr int Hh  = 512;    // hidden
static constexpr int Nn  = 128;    // memory slots
static constexpr int WD_ = 64;     // word size
static constexpr int R_  = 4;      // read heads
static constexpr int G4  = 4 * Hh; // 2048 gates
static constexpr int KIN = 512;    // E + R*WD
static constexpr int IF_ = 471;    // iface size
static constexpr int IFP = 480;    // iface padded to x16
static constexpr int HC  = Hh + R_ * WD_; // 768 concat for out proj
static constexpr float EPSC = 1e-6f;

typedef __bf16 bf16_t;
typedef __attribute__((ext_vector_type(16))) __bf16 v16bf;
typedef __attribute__((ext_vector_type(8)))  float  v8f;
typedef __attribute__((ext_vector_type(4)))  int    v4i;

#define AS1 __attribute__((address_space(1)))
#define AS3 __attribute__((address_space(3)))

// ---------------- CDNA5 async global->LDS staging ----------------
#if __has_builtin(__builtin_amdgcn_global_load_async_to_lds_b128) && \
    __has_builtin(__builtin_amdgcn_s_wait_asynccnt)
#define USE_ASYNC_LDS 1
#else
#define USE_ASYNC_LDS 0
#endif

__device__ inline void async_copy_b128(void* lds, const void* g) {
#if USE_ASYNC_LDS
  // prototype (from compiler diagnostic): (v4i AS1*, v4i AS3*, imm offset, imm cpol)
  __builtin_amdgcn_global_load_async_to_lds_b128((AS1 v4i*)(void*)g, (AS3 v4i*)lds, 0, 0);
#else
  *(uint4*)lds = *(const uint4*)g;
#endif
}
__device__ inline void async_wait0() {
#if USE_ASYNC_LDS
  __builtin_amdgcn_s_wait_asynccnt(0);
#endif
}

// ---------------- WMMA helpers (CDNA5 gfx1250, wave32) ----------------
__device__ inline v8f wmma_bf16(v16bf a, v16bf b, v8f c) {
  // 8 args: (neg_a, A, neg_b, B, c_mod, C, reuse_a, reuse_b)
  return __builtin_amdgcn_wmma_f32_16x16x32_bf16(false, a, false, b, (short)0, c, false, false);
}

// 16-bit A-matrix 16x32 fragment: lane L row m=L&15; lanes<16 hold K{k0..k0+7,k0+16..k0+23},
// lanes>=16 hold K{k0+8..k0+15,k0+24..k0+31}.
__device__ inline v16bf load_a(const bf16_t* A, int ld, int k0, int lane) {
  const bf16_t* row = A + (size_t)(lane & 15) * ld + k0 + ((lane >> 4) << 3);
  v16bf a;
#pragma unroll
  for (int i = 0; i < 8; ++i) a[i] = row[i];
#pragma unroll
  for (int i = 0; i < 8; ++i) a[8 + i] = row[16 + i];
  return a;
}

// Pre-packed B fragment: per (tile,kb) a 32x16-element block, lane-major, 16 contiguous per lane.
__device__ inline const bf16_t* bblk(const bf16_t* base, int tile, int kb, int kBlks) {
  return base + (((size_t)tile * kBlks + kb) << 9); // *512 elems
}
__device__ inline v16bf load_b(const bf16_t* blk, int lane) {
  const bf16_t* p = blk + lane * 16;
  v16bf b;
#pragma unroll
  for (int i = 0; i < 16; ++i) b[i] = p[i];
  return b;
}

// ---------------- small math helpers ----------------
__device__ inline float sigf(float x) { return 1.f / (1.f + expf(-x)); }
__device__ inline float softplusf(float x) { return x > 20.f ? x : log1pf(expf(x)); }
__device__ inline float wave_sum(float v) {
#pragma unroll
  for (int o = 16; o > 0; o >>= 1) v += __shfl_xor(v, o, 32);
  return v;
}
__device__ inline float wave_max(float v) {
#pragma unroll
  for (int o = 16; o > 0; o >>= 1) v = fmaxf(v, __shfl_xor(v, o, 32));
  return v;
}

// ---------------- prep kernels ----------------
// Pack weight matrix into WMMA B-fragment layout (bf16).
// transpose=1: B[k][n] = src[n*K + k]   (src is (Nsrc x K), e.g. w_ih)
// transpose=0: B[k][n] = src[k*Nsrc + n] (src is (K x Nsrc), e.g. W_out)
__global__ void pack_b_kernel(const float* __restrict__ src, bf16_t* __restrict__ dst,
                              int K, int Nsrc, int nTiles, int transpose) {
  int idx = blockIdx.x * blockDim.x + threadIdx.x;
  int kBlks = K >> 5;
  int total = nTiles * kBlks * 32;
  if (idx >= total) return;
  int lane = idx & 31;
  int kb   = (idx >> 5) % kBlks;
  int tile = (idx >> 5) / kBlks;
  int n = tile * 16 + (lane & 15);
  int kbase = kb * 32 + (lane >> 4) * 8;
  bf16_t* out = dst + (((size_t)tile * kBlks + kb) << 9) + (size_t)lane * 16;
#pragma unroll
  for (int i = 0; i < 16; ++i) {
    int k = kbase + (i < 8 ? i : i + 8);
    float v = 0.f;
    if (n < Nsrc) v = transpose ? src[(size_t)n * K + k] : src[(size_t)k * Nsrc + n];
    out[i] = (bf16_t)v;
  }
}

__global__ void embed_kernel(const int* __restrict__ tokens, const float* __restrict__ emb,
                             bf16_t* __restrict__ xbf, int total) {
  int idx = blockIdx.x * blockDim.x + threadIdx.x;
  if (idx >= total) return;
  int e = idx & (E_ - 1);
  int bt = idx >> 8; // /E_
  int tok = tokens[bt];
  xbf[idx] = (bf16_t)tanhf(emb[(size_t)tok * E_ + e]);
}

__global__ void zero_kernel(float* __restrict__ p, size_t n) {
  size_t i = (size_t)blockIdx.x * blockDim.x + threadIdx.x;
  if (i < n) p[i] = 0.f;
}

// ---------------- the recurrent scan ----------------
// grid = 2 (16 batch rows each), block = 512 threads = 16 wave32s.
__global__ __launch_bounds__(512)
void dnc_scan_kernel(const bf16_t* __restrict__ xbf,
                     const bf16_t* __restrict__ pkIH,
                     const bf16_t* __restrict__ pkHH,
                     const bf16_t* __restrict__ pkIF,
                     const float* __restrict__ b_lstm,
                     const float* __restrict__ b_iface,
                     float* __restrict__ gbuf,
                     float* stateH, float* stateC, float* stateM, float* stateL,
                     float* stateP, float* stateU, float* stateWr, float* stateWw,
                     float* stateRv,
                     bf16_t* __restrict__ hcat) {
  __shared__ __align__(16) bf16_t s_inp[16 * 512];  // [x_t | rv] bf16
  __shared__ __align__(16) bf16_t s_h[16 * 512];    // h bf16
  __shared__ float  s_xi[16 * IFP];                 // iface outputs fp32

  const int tid  = threadIdx.x;
  const int wave = tid >> 5;
  const int lane = tid & 31;
  const int bbase = blockIdx.x * 16;
  float* g = gbuf + (size_t)blockIdx.x * 16 * G4;

  for (int t = 0; t < T_; ++t) {
    // ---- Phase 1: stage inputs & h into LDS (bf16) ----
    // x_t part (16 rows x 256 cols bf16, 16B chunks) via async global->LDS DMA.
    {
      int m   = tid >> 5;        // 0..15 row
      int seg = tid & 31;        // 0..31 -> 8 bf16 each = 256 cols
      int b = bbase + m;
      async_copy_b128(&s_inp[m * 512 + seg * 8],
                      &xbf[((size_t)b * T_ + t) * E_ + seg * 8]);
    }
    // rv part + h staging (fp32 -> bf16 conversion, plain path)
    for (int idx = tid; idx < 16 * 256; idx += 512) {
      int m = idx >> 8, j = idx & 255;
      int b = bbase + m;
      s_inp[m * 512 + 256 + j] = (bf16_t)stateRv[(size_t)b * (R_ * WD_) + j];
    }
    for (int idx = tid; idx < 16 * 512; idx += 512) {
      int m = idx >> 9, j = idx & 511;
      int b = bbase + m;
      s_h[idx] = (bf16_t)stateH[(size_t)b * Hh + j];
    }
    async_wait0();
    __syncthreads();

    // ---- Phase 2: gates = inp@w_ih^T + h@w_hh^T + b  (WMMA, 16x2048) ----
    for (int tile = wave; tile < G4 / 16; tile += 16) {
      v8f acc = {0.f, 0.f, 0.f, 0.f, 0.f, 0.f, 0.f, 0.f};
      for (int kb = 0; kb < KIN / 32; ++kb) {
        v16bf a  = load_a(s_inp, 512, kb * 32, lane);
        v16bf bb = load_b(bblk(pkIH, tile, kb, KIN / 32), lane);
        acc = wmma_bf16(a, bb, acc);
        v16bf a2 = load_a(s_h, 512, kb * 32, lane);
        v16bf b2 = load_b(bblk(pkHH, tile, kb, KIN / 32), lane);
        acc = wmma_bf16(a2, b2, acc);
      }
      int n = tile * 16 + (lane & 15);
      float bias = b_lstm[n];
      int mb = (lane >> 4) * 8;
#pragma unroll
      for (int rr = 0; rr < 8; ++rr) g[(size_t)(mb + rr) * G4 + n] = acc[rr] + bias;
    }
    __threadfence();
    __syncthreads();

    // ---- Phase 3: LSTM cell elementwise; write new h to LDS/global/hcat ----
    for (int idx = tid; idx < 16 * 512; idx += 512) {
      int m = idx >> 9, j = idx & 511;
      int b = bbase + m;
      float gi = g[(size_t)m * G4 + j];
      float gf = g[(size_t)m * G4 + 512 + j];
      float gg = g[(size_t)m * G4 + 1024 + j];
      float go = g[(size_t)m * G4 + 1536 + j];
      float c = sigf(gf) * stateC[(size_t)b * Hh + j] + sigf(gi) * tanhf(gg);
      float h = sigf(go) * tanhf(c);
      stateC[(size_t)b * Hh + j] = c;
      stateH[(size_t)b * Hh + j] = h;
      s_h[idx] = (bf16_t)h;
      hcat[((size_t)b * T_ + t) * HC + j] = (bf16_t)h;
    }
    __syncthreads();

    // ---- Phase 4: xi = h @ W_iface + b  (WMMA, 16x480) ----
    for (int tile = wave; tile < IFP / 16; tile += 16) {
      v8f acc = {0.f, 0.f, 0.f, 0.f, 0.f, 0.f, 0.f, 0.f};
      for (int kb = 0; kb < KIN / 32; ++kb) {
        v16bf a  = load_a(s_h, 512, kb * 32, lane);
        v16bf bb = load_b(bblk(pkIF, tile, kb, KIN / 32), lane);
        acc = wmma_bf16(a, bb, acc);
      }
      int n = tile * 16 + (lane & 15);
      float bias = (n < IF_) ? b_iface[n] : 0.f;
      int mb = (lane >> 4) * 8;
#pragma unroll
      for (int rr = 0; rr < 8; ++rr) s_xi[(size_t)(mb + rr) * IFP + n] = acc[rr] + bias;
    }
    __syncthreads();

    // ---- Phase 5: DNC memory ops. One wave per batch row, N=128 => 4 slots/lane ----
    {
      const int mrow = wave;
      const int b = bbase + mrow;
      const float* xi = &s_xi[(size_t)mrow * IFP];
      float* M  = stateM + (size_t)b * (Nn * WD_);
      float* L  = stateL + (size_t)b * (Nn * Nn);
      float* P  = stateP + (size_t)b * Nn;
      float* U  = stateU + (size_t)b * Nn;
      float* WR = stateWr + (size_t)b * (R_ * Nn);
      float* WW = stateWw + (size_t)b * Nn;
      float* RV = stateRv + (size_t)b * (R_ * WD_);

      // interface scalars (xi offsets: rk 0, rb 256, wk 260, wb 324, er 325, wv 389,
      //                    fg 453, ga 457, gw 458, pi 459)
      float rb_[4], fg_[4], pi_[4][3];
#pragma unroll
      for (int r = 0; r < 4; ++r) {
        rb_[r] = 1.f + softplusf(xi[256 + r]);
        fg_[r] = sigf(xi[453 + r]);
      }
      float wb_ = 1.f + softplusf(xi[324]);
      float ga_ = sigf(xi[457]);
      float gw_ = sigf(xi[458]);
#pragma unroll
      for (int r = 0; r < 4; ++r) {
        float e0 = xi[459 + r * 3 + 0], e1 = xi[459 + r * 3 + 1], e2 = xi[459 + r * 3 + 2];
        float mx = fmaxf(e0, fmaxf(e1, e2));
        float x0 = expf(e0 - mx), x1 = expf(e1 - mx), x2 = expf(e2 - mx);
        float s = x0 + x1 + x2;
        pi_[r][0] = x0 / s; pi_[r][1] = x1 / s; pi_[r][2] = x2 / s;
      }

      // load old per-slot state (lane owns n = lane + 32*q)
      float wrold[4][4], uo[4], po[4], wwo[4];
#pragma unroll
      for (int q = 0; q < 4; ++q) {
        int n = lane + 32 * q;
        uo[q] = U[n]; po[q] = P[n]; wwo[q] = WW[n];
#pragma unroll
        for (int r = 0; r < 4; ++r) wrold[r][q] = WR[r * Nn + n];
      }

      // retention psi and usage u
      float u_[4];
#pragma unroll
      for (int q = 0; q < 4; ++q) {
        float psi = 1.f;
#pragma unroll
        for (int r = 0; r < 4; ++r) psi *= (1.f - fg_[r] * wrold[r][q]);
        u_[q] = (uo[q] + wwo[q] - uo[q] * wwo[q]) * psi;
      }

      // allocation weights: a[n] = (1-u[n]) * prod_{(u[j],j)<(u[n],n)} u[j]
      float prod[4] = {1.f, 1.f, 1.f, 1.f};
#pragma unroll
      for (int jq = 0; jq < 4; ++jq)
        for (int jl = 0; jl < 32; ++jl) {
          int j = jl + 32 * jq;
          float uj = __shfl(u_[jq], jl, 32);
#pragma unroll
          for (int q = 0; q < 4; ++q) {
            int n = lane + 32 * q;
            if (uj < u_[q] || (uj == u_[q] && j < n)) prod[q] *= uj;
          }
        }
      float al[4];
#pragma unroll
      for (int q = 0; q < 4; ++q) al[q] = (1.f - u_[q]) * prod[q];

      // write content weighting (cosine with OLD M)
      float t0;
      { float w0 = xi[260 + lane], w1 = xi[260 + lane + 32]; t0 = w0 * w0 + w1 * w1; }
      float kn = sqrtf(wave_sum(t0));
      float cs[4];
#pragma unroll
      for (int q = 0; q < 4; ++q) {
        int n = lane + 32 * q;
        const float* Mr = M + (size_t)n * WD_;
        float dot = 0.f, nn = 0.f;
        for (int w = 0; w < WD_; ++w) { float mv = Mr[w]; dot += mv * xi[260 + w]; nn += mv * mv; }
        cs[q] = wb_ * dot / (kn * sqrtf(nn) + EPSC);
      }
      float mx = wave_max(fmaxf(fmaxf(cs[0], cs[1]), fmaxf(cs[2], cs[3])));
      float cw[4]; float ssum = 0.f;
#pragma unroll
      for (int q = 0; q < 4; ++q) { cw[q] = expf(cs[q] - mx); ssum += cw[q]; }
      ssum = wave_sum(ssum);
#pragma unroll
      for (int q = 0; q < 4; ++q) cw[q] /= ssum;

      // write weights
      float ww_[4];
#pragma unroll
      for (int q = 0; q < 4; ++q) ww_[q] = gw_ * (ga_ * al[q] + (1.f - ga_) * cw[q]);

      // memory update (cache new row sq-norms)
      float Mn2[4];
#pragma unroll
      for (int q = 0; q < 4; ++q) {
        int n = lane + 32 * q;
        float* Mr = M + (size_t)n * WD_;
        float wwn = ww_[q]; float nn = 0.f;
        for (int w = 0; w < WD_; ++w) {
          float er = sigf(xi[325 + w]);
          float wv = xi[389 + w];
          float mv = Mr[w];
          mv = mv * (1.f - wwn * er) + wwn * wv;
          Mr[w] = mv; nn += mv * mv;
        }
        Mn2[q] = nn;
      }

      // temporal link matrix update (uses old p)
      float wwsum = wave_sum(ww_[0] + ww_[1] + ww_[2] + ww_[3]);
#pragma unroll
      for (int jq = 0; jq < 4; ++jq)
        for (int jl = 0; jl < 32; ++jl) {
          int j = jl + 32 * jq;
          float wwj = __shfl(ww_[jq], jl, 32);
          float pj  = __shfl(po[jq], jl, 32);
#pragma unroll
          for (int q = 0; q < 4; ++q) {
            int n = lane + 32 * q;
            float v = (n == j) ? 0.f
                               : (1.f - ww_[q] - wwj) * L[(size_t)n * Nn + j] + ww_[q] * pj;
            L[(size_t)n * Nn + j] = v;
          }
        }
      float p_[4];
#pragma unroll
      for (int q = 0; q < 4; ++q) p_[q] = (1.f - wwsum) * po[q] + ww_[q];

      __threadfence(); // make this wave's M/L stores visible across its own lanes

      // forward/backward weights + content read + new read weights
      float wrnew[4][4];
#pragma unroll
      for (int r = 0; r < 4; ++r) {
        float fw[4] = {0.f, 0.f, 0.f, 0.f};
        float bw[4] = {0.f, 0.f, 0.f, 0.f};
#pragma unroll
        for (int jq = 0; jq < 4; ++jq)
          for (int jl = 0; jl < 32; ++jl) {
            int mj = jl + 32 * jq;
            float wrm = __shfl(wrold[r][jq], jl, 32);
#pragma unroll
            for (int q = 0; q < 4; ++q) {
              int n = lane + 32 * q;
              fw[q] += L[(size_t)n * Nn + mj] * wrm;
              bw[q] += L[(size_t)mj * Nn + n] * wrm;
            }
          }
        float tt;
        { float k0 = xi[r * 64 + lane], k1 = xi[r * 64 + lane + 32]; tt = k0 * k0 + k1 * k1; }
        float knr = sqrtf(wave_sum(tt));
        float csr[4];
#pragma unroll
        for (int q = 0; q < 4; ++q) {
          int n = lane + 32 * q;
          const float* Mr = M + (size_t)n * WD_;
          float dot = 0.f;
          for (int w = 0; w < WD_; ++w) dot += Mr[w] * xi[r * 64 + w];
          csr[q] = rb_[r] * dot / (knr * sqrtf(Mn2[q]) + EPSC);
        }
        float mxr = wave_max(fmaxf(fmaxf(csr[0], csr[1]), fmaxf(csr[2], csr[3])));
        float crq[4]; float sr = 0.f;
#pragma unroll
        for (int q = 0; q < 4; ++q) { crq[q] = expf(csr[q] - mxr); sr += crq[q]; }
        sr = wave_sum(sr);
#pragma unroll
        for (int q = 0; q < 4; ++q) {
          crq[q] /= sr;
          wrnew[r][q] = pi_[r][0] * bw[q] + pi_[r][1] * crq[q] + pi_[r][2] * fw[q];
        }
      }

      // read vectors rv = wr_new @ M_new ; write to state + hcat(bf16)
#pragma unroll
      for (int r = 0; r < 4; ++r) {
        float a0 = 0.f, a1 = 0.f;
#pragma unroll
        for (int jq = 0; jq < 4; ++jq)
          for (int jl = 0; jl < 32; ++jl) {
            int n = jl + 32 * jq;
            float wn = __shfl(wrnew[r][jq], jl, 32);
            const float* Mr = M + (size_t)n * WD_;
            a0 += wn * Mr[lane];
            a1 += wn * Mr[lane + 32];
          }
        RV[r * WD_ + lane] = a0;
        RV[r * WD_ + lane + 32] = a1;
        bf16_t* hc = hcat + ((size_t)b * T_ + t) * HC + Hh + r * WD_;
        hc[lane] = (bf16_t)a0;
        hc[lane + 32] = (bf16_t)a1;
      }
#pragma unroll
      for (int q = 0; q < 4; ++q) {
        int n = lane + 32 * q;
        U[n] = u_[q]; P[n] = p_[q]; WW[n] = ww_[q];
#pragma unroll
        for (int r = 0; r < 4; ++r) WR[r * Nn + n] = wrnew[r][q];
      }
    }
    __threadfence();
    __syncthreads();
  }
}

// ---------------- output projection: (8192 x 768) @ (768 x 10000) ----------------
// grid = (512 mTiles, 20 nGroups), block = 256 (8 waves); each wave does a 16x64 strip.
// A-panel (16x768 bf16 = 24KB) staged once per block via async global->LDS DMA.
__global__ __launch_bounds__(256)
void outproj_kernel(const bf16_t* __restrict__ hcat, const bf16_t* __restrict__ pkOUT,
                    const float* __restrict__ b_out, float* __restrict__ out) {
  __shared__ __align__(16) bf16_t sA[16 * HC]; // 24KB A panel
  const int wave = threadIdx.x >> 5;
  const int lane = threadIdx.x & 31;
  const int mt = blockIdx.x;
  const int t0 = blockIdx.y * 32 + wave * 4;
  const int nTiles = V_ / 16; // 625
  const int kBlks = HC / 32;  // 24
  const bf16_t* A = hcat + (size_t)mt * 16 * HC;

  // stage A panel: 12288 bf16 = 1536 x b128, 6 per thread
  for (int o = threadIdx.x * 8; o < 16 * HC; o += 256 * 8)
    async_copy_b128(&sA[o], &A[o]);
  async_wait0();
  __syncthreads();

  v8f acc[4];
#pragma unroll
  for (int j = 0; j < 4; ++j) acc[j] = (v8f){0.f, 0.f, 0.f, 0.f, 0.f, 0.f, 0.f, 0.f};

  for (int kb = 0; kb < kBlks; ++kb) {
    if (kb + 1 < kBlks && t0 < nTiles)
      __builtin_prefetch(bblk(pkOUT, t0, kb + 1, kBlks), 0, 0);
    v16bf a = load_a(sA, HC, kb * 32, lane);
#pragma unroll
    for (int j = 0; j < 4; ++j) {
      int tile = t0 + j;
      if (tile < nTiles) {
        v16bf bb = load_b(bblk(pkOUT, tile, kb, kBlks), lane);
        acc[j] = wmma_bf16(a, bb, acc[j]);
      }
    }
  }
  int mb = (lane >> 4) * 8;
#pragma unroll
  for (int j = 0; j < 4; ++j) {
    int tile = t0 + j;
    if (tile < nTiles) {
      int n = tile * 16 + (lane & 15);
      float bias = b_out[n];
#pragma unroll
      for (int rr = 0; rr < 8; ++rr)
        out[(size_t)(mt * 16 + mb + rr) * V_ + n] = acc[j][rr] + bias;
    }
  }
}

// ---------------- host launch ----------------
extern "C" void kernel_launch(void* const* d_in, const int* in_sizes, int n_in,
                              void* d_out, int out_size, void* d_ws, size_t ws_size,
                              hipStream_t stream) {
  (void)in_sizes; (void)n_in; (void)out_size; (void)ws_size;
  const int*   tokens  = (const int*)d_in[0];
  const float* emb     = (const float*)d_in[1];
  const float* w_ih    = (const float*)d_in[2];
  const float* w_hh    = (const float*)d_in[3];
  const float* b_lstm  = (const float*)d_in[4];
  const float* W_iface = (const float*)d_in[5];
  const float* b_iface = (const float*)d_in[6];
  const float* W_out   = (const float*)d_in[7];
  const float* b_out   = (const float*)d_in[8];
  float* out = (float*)d_out;

  // workspace carve
  char* ws = (char*)d_ws;
  size_t off = 0;
  auto carve = [&](size_t bytes) -> char* {
    char* p = ws + off;
    off += (bytes + 255) & ~(size_t)255;
    return p;
  };
  bf16_t* xbf   = (bf16_t*)carve((size_t)Bc * T_ * E_ * 2);
  bf16_t* pkIH  = (bf16_t*)carve((size_t)(G4 / 16) * (KIN / 32) * 512 * 2);
  bf16_t* pkHH  = (bf16_t*)carve((size_t)(G4 / 16) * (KIN / 32) * 512 * 2);
  bf16_t* pkIF  = (bf16_t*)carve((size_t)(IFP / 16) * (KIN / 32) * 512 * 2);
  bf16_t* pkOUT = (bf16_t*)carve((size_t)(V_ / 16) * (HC / 32) * 512 * 2);
  bf16_t* hcat  = (bf16_t*)carve((size_t)Bc * T_ * HC * 2);
  float*  gbuf  = (float*)carve((size_t)2 * 16 * G4 * 4);
  // contiguous state block (zeroed once per launch)
  const size_t fH = (size_t)Bc * Hh, fC = fH;
  const size_t fM = (size_t)Bc * Nn * WD_, fL = (size_t)Bc * Nn * Nn;
  const size_t fP = (size_t)Bc * Nn, fU = fP, fWW = fP;
  const size_t fWR = (size_t)Bc * R_ * Nn, fRV = (size_t)Bc * R_ * WD_;
  const size_t stateFloats = fH + fC + fM + fL + fP + fU + fWR + fWW + fRV;
  float* stateAll = (float*)carve(stateFloats * 4);
  float* sH = stateAll;
  float* sC = sH + fH;
  float* sM = sC + fC;
  float* sL = sM + fM;
  float* sP = sL + fL;
  float* sU = sP + fP;
  float* sWR = sU + fU;
  float* sWW = sWR + fWR;
  float* sRV = sWW + fWW;

  // --- prep: pack weights to WMMA B-fragment layout (bf16) ---
  {
    int th = (G4 / 16) * (KIN / 32) * 32; // 65536
    pack_b_kernel<<<(th + 255) / 256, 256, 0, stream>>>(w_ih, pkIH, KIN, G4, G4 / 16, 1);
    pack_b_kernel<<<(th + 255) / 256, 256, 0, stream>>>(w_hh, pkHH, KIN, G4, G4 / 16, 1);
  }
  {
    int th = (IFP / 16) * (KIN / 32) * 32; // 15360
    pack_b_kernel<<<(th + 255) / 256, 256, 0, stream>>>(W_iface, pkIF, KIN, IF_, IFP / 16, 0);
  }
  {
    int th = (V_ / 16) * (HC / 32) * 32; // 480000
    pack_b_kernel<<<(th + 255) / 256, 256, 0, stream>>>(W_out, pkOUT, HC, V_, V_ / 16, 0);
  }
  // --- prep: embeddings (tanh) ---
  {
    int th = Bc * T_ * E_;
    embed_kernel<<<(th + 255) / 256, 256, 0, stream>>>(tokens, emb, xbf, th);
  }
  // --- prep: zero recurrent state ---
  zero_kernel<<<(unsigned)((stateFloats + 255) / 256), 256, 0, stream>>>(stateAll, stateFloats);

  // --- recurrent scan (2 WGs x 16 waves) ---
  dnc_scan_kernel<<<2, 512, 0, stream>>>(xbf, pkIH, pkHH, pkIF, b_lstm, b_iface, gbuf,
                                         sH, sC, sM, sL, sP, sU, sWR, sWW, sRV, hcat);

  // --- batched output projection ---
  dim3 grid((Bc * T_) / 16, (V_ / 16 + 31) / 32); // 512 x 20
  outproj_kernel<<<grid, 256, 0, stream>>>(hcat, pkOUT, b_out, out);
}
